// SelfAttention_16621523435582
// MI455X (gfx1250) — matrix operand order repaired and checked by reading, not compiled, not run on hardware
//
#include <hip/hip_runtime.h>
#include <hip/hip_bf16.h>

typedef _Float16 v16h __attribute__((ext_vector_type(16)));
typedef _Float16 v8h  __attribute__((ext_vector_type(8)));
typedef _Float16 v4h  __attribute__((ext_vector_type(4)));
typedef float    v8f  __attribute__((ext_vector_type(8)));

#define D_MODEL  512
#define N_HEADS  8
#define HEAD_DIM 64
#define BATCH    2
#define SEQ      4096
#define M_TOTAL  (BATCH*SEQ)      /* 8192 */
#define N_QKV    (3*D_MODEL)      /* 1536 */

#define WMMA_F16(a,b,c) __builtin_amdgcn_wmma_f32_16x16x32_f16(false,(a),false,(b),(short)0,(c),false,false)

// ---- A-matrix (16x32 f16): lane holds row M=lane%16; element e -> k = kk + 16*half + e
static __device__ __forceinline__ v16h wmma_a_f16(const _Float16* rowp, int kk, int half) {
    const v8h c0 = *(const v8h*)(rowp + kk + half * 16);
    const v8h c1 = *(const v8h*)(rowp + kk + half * 16 + 8);
    v16h a;
#pragma unroll
    for (int e = 0; e < 8; ++e) { a[e] = c0[e]; a[e + 8] = c1[e]; }
    return a;
}
// ---- B-matrix (32x16 f16): lane holds col N=lane%16; element e -> k = kk + 16*half + e
// A transposed-weight / K-row / V^T-row is exactly one contiguous 32B register.
static __device__ __forceinline__ v16h wmma_b_contig(const _Float16* p) {
    return *(const v16h*)p;
}

// cross-lane reductions within 16-lane half-groups (wave32; xor masks <=8 stay in-group)
static __device__ __forceinline__ float rmax16(float v) {
    v = fmaxf(v, __shfl_xor(v, 8, 32));
    v = fmaxf(v, __shfl_xor(v, 4, 32));
    v = fmaxf(v, __shfl_xor(v, 2, 32));
    v = fmaxf(v, __shfl_xor(v, 1, 32));
    return v;
}
static __device__ __forceinline__ float rsum16(float v) {
    v += __shfl_xor(v, 8, 32);
    v += __shfl_xor(v, 4, 32);
    v += __shfl_xor(v, 2, 32);
    v += __shfl_xor(v, 1, 32);
    return v;
}

// ===================== Prep: fp32 -> f16 straight copy (x) =====================
__global__ __launch_bounds__(256) void k_cvt(const float* __restrict__ in,
                                             _Float16* __restrict__ out) {
    const size_t i = ((size_t)blockIdx.x * 256 + threadIdx.x) * 4;
    const float4 f = *(const float4*)(in + i);
    v4h h; h[0] = (_Float16)f.x; h[1] = (_Float16)f.y; h[2] = (_Float16)f.z; h[3] = (_Float16)f.w;
    *(v4h*)(out + i) = h;
}
// ===================== Prep: fp32 [K x N] -> f16 transposed [N x K] =====================
__global__ __launch_bounds__(256) void k_cvt_wT(const float* __restrict__ in,
                                                _Float16* __restrict__ out,
                                                int K, int N) {
    const size_t idx = (size_t)blockIdx.x * 256 + threadIdx.x;
    const int n = (int)(idx / K), k = (int)(idx % K);
    out[idx] = (_Float16)in[(size_t)k * N + n];
}

// ===================== Kernel 1: QKV projection (32x64 per wave) =====================
// x16[8192x512] @ WqT(f16,[1536x512]) + b -> Q,K f16 [B,H,S,hd]; V stored transposed [B,H,hd,S]
__global__ __launch_bounds__(256) void k_qkv(const _Float16* __restrict__ x16,
                                             const _Float16* __restrict__ WqT,
                                             const float* __restrict__ bias,
                                             _Float16* __restrict__ qb,
                                             _Float16* __restrict__ kb,
                                             _Float16* __restrict__ vtb) {
    const int lane = threadIdx.x & 31, w = threadIdx.x >> 5;
    const int half = lane >> 4, n = lane & 15;
    const int NT = N_QKV / 64;                 // 24 col tiles of 64
    const int tile = blockIdx.x * 8 + w;
    const int tm = tile / NT, tn = tile % NT;

    const _Float16* xrow0 = x16 + (size_t)(tm * 32 + n) * D_MODEL;
    const _Float16* xrow1 = xrow0 + (size_t)16 * D_MODEL;
    const _Float16* wbase = WqT + (size_t)(tn * 64 + n) * D_MODEL + 16 * half;

    v8f acc[2][4];
#pragma unroll
    for (int i = 0; i < 2; ++i)
#pragma unroll
        for (int t = 0; t < 4; ++t) acc[i][t] = (v8f){};

    for (int kk = 0; kk < D_MODEL; kk += 32) {
        const v16h a0 = wmma_a_f16(xrow0, kk, half);
        const v16h a1 = wmma_a_f16(xrow1, kk, half);
#pragma unroll
        for (int t = 0; t < 4; ++t) {
            const v16h b = wmma_b_contig(wbase + (size_t)(t * 16) * D_MODEL + kk);
            acc[0][t] = WMMA_F16(a0, b, acc[0][t]);
            acc[1][t] = WMMA_F16(a1, b, acc[1][t]);
        }
    }

    const int which = (tn * 64) >> 9;          // 0=Q 1=K 2=V (uniform per wave)
    const int h = ((tn * 64) & 511) >> 6;      // head (uniform: 64-wide tile within one head)
#pragma unroll
    for (int t = 0; t < 4; ++t) {
        const int col = tn * 64 + t * 16 + n;
        const float bv = bias[col];
        const int hd = t * 16 + n;
        if (which == 2) {
#pragma unroll
            for (int i = 0; i < 2; ++i)
#pragma unroll
                for (int r = 0; r < 8; ++r) {
                    const int m = tm * 32 + i * 16 + r + 8 * half;
                    const int b_ = m >> 12, s = m & 4095;
                    vtb[((size_t)(b_ * N_HEADS + h) * HEAD_DIM + hd) * SEQ + s] =
                        (_Float16)(acc[i][t][r] + bv);
                }
        } else {
            _Float16* dst = (which == 0) ? qb : kb;
#pragma unroll
            for (int i = 0; i < 2; ++i)
#pragma unroll
                for (int r = 0; r < 8; ++r) {
                    const int m = tm * 32 + i * 16 + r + 8 * half;
                    const int b_ = m >> 12, s = m & 4095;
                    dst[(((size_t)(b_ * N_HEADS + h)) * SEQ + s) * HEAD_DIM + hd] =
                        (_Float16)(acc[i][t][r] + bv);
                }
        }
    }
}

// ===================== Kernel 2: causal flash attention =====================
// One wave = one 16-row Q tile; 64-key blocks, online softmax, 16 WMMAs per block.
__global__ __launch_bounds__(256) void k_attn(const _Float16* __restrict__ qb,
                                              const _Float16* __restrict__ kb,
                                              const _Float16* __restrict__ vtb,
                                              _Float16* __restrict__ ab) {
    __shared__ _Float16 lds_p[8][16][72];      // per-wave P staging; rows padded to 144B
    const int lane = threadIdx.x & 31, w = threadIdx.x >> 5;
    const int half = lane >> 4, n = lane & 15;
    const int QT = SEQ / 16;
    const int gt = blockIdx.x * 8 + w;
    const int bh = gt / QT, qt = gt % QT;
    const int q0 = qt * 16;

    const _Float16* Q  = qb  + (size_t)bh * SEQ * HEAD_DIM;
    const _Float16* K  = kb  + (size_t)bh * SEQ * HEAD_DIM;
    const _Float16* VT = vtb + (size_t)bh * HEAD_DIM * SEQ;   // [hd][S]

    const _Float16* qrow = Q + (size_t)(q0 + n) * HEAD_DIM;
    const v16h aq0 = wmma_a_f16(qrow, 0, half);
    const v16h aq1 = wmma_a_f16(qrow, 32, half);

    v8f acc0 = {}, acc1 = {}, acc2 = {}, acc3 = {};
    float mrun[8], lrun[8];
#pragma unroll
    for (int r = 0; r < 8; ++r) { mrun[r] = -1e30f; lrun[r] = 0.0f; }

    const int njb = (qt + 4) >> 2;             // 64-key blocks covering [0, q0+16)
    for (int jb = 0; jb < njb; ++jb) {
        const int j0 = jb * 64;
        if (jb + 1 < njb) {                    // prefetch next K/V^T tiles (global_prefetch_b8)
            __builtin_prefetch(K + (size_t)(j0 + 64 + lane) * HEAD_DIM, 0, 0);
            __builtin_prefetch(VT + (size_t)(lane * 2) * SEQ + j0 + 64, 0, 0);
        }
        // ---- scores: s[c] covers keys j0+16c .. j0+16c+15
        v8f s0 = {}, s1 = {}, s2 = {}, s3 = {};
        {
            const _Float16* kr = K + (size_t)(j0 + n) * HEAD_DIM + 16 * half;
            s0 = WMMA_F16(aq0, wmma_b_contig(kr), s0);
            s0 = WMMA_F16(aq1, wmma_b_contig(kr + 32), s0);
            kr += 16 * HEAD_DIM;
            s1 = WMMA_F16(aq0, wmma_b_contig(kr), s1);
            s1 = WMMA_F16(aq1, wmma_b_contig(kr + 32), s1);
            kr += 16 * HEAD_DIM;
            s2 = WMMA_F16(aq0, wmma_b_contig(kr), s2);
            s2 = WMMA_F16(aq1, wmma_b_contig(kr + 32), s2);
            kr += 16 * HEAD_DIM;
            s3 = WMMA_F16(aq0, wmma_b_contig(kr), s3);
            s3 = WMMA_F16(aq1, wmma_b_contig(kr + 32), s3);
        }
        // ---- scale + causal mask + online softmax (rows r+8*half, cols across lanes)
#pragma unroll
        for (int r = 0; r < 8; ++r) {
            const int qr = q0 + r + 8 * half;
            float v0 = s0[r] * 0.125f; if (j0 + n      > qr) v0 = -1e30f;
            float v1 = s1[r] * 0.125f; if (j0 + 16 + n > qr) v1 = -1e30f;
            float v2 = s2[r] * 0.125f; if (j0 + 32 + n > qr) v2 = -1e30f;
            float v3 = s3[r] * 0.125f; if (j0 + 48 + n > qr) v3 = -1e30f;
            const float rowmax = rmax16(fmaxf(fmaxf(v0, v1), fmaxf(v2, v3)));
            const float nm = fmaxf(mrun[r], rowmax);
            const float alpha = __expf(mrun[r] - nm);
            const float p0 = __expf(v0 - nm);
            const float p1 = __expf(v1 - nm);
            const float p2 = __expf(v2 - nm);
            const float p3 = __expf(v3 - nm);
            const float rs = rsum16((p0 + p1) + (p2 + p3));
            lrun[r] = lrun[r] * alpha + rs;
            mrun[r] = nm;
            acc0[r] *= alpha; acc1[r] *= alpha; acc2[r] *= alpha; acc3[r] *= alpha;
            _Float16* pr = &lds_p[w][r + 8 * half][n];
            pr[0] = (_Float16)p0; pr[16] = (_Float16)p1; pr[32] = (_Float16)p2; pr[48] = (_Float16)p3;
        }
        asm volatile("s_wait_dscnt 0x0" ::: "memory");   // cross-lane LDS RAW inside the wave
        // ---- reload P in A-layout: pa0 = k 0..31, pa1 = k 32..63
        v16h pa0, pa1;
        {
            const _Float16* pr = &lds_p[w][n][0];
            const v8h c0 = *(const v8h*)(pr + half * 16);
            const v8h c1 = *(const v8h*)(pr + 8 + half * 16);
            const v8h c2 = *(const v8h*)(pr + 32 + half * 16);
            const v8h c3 = *(const v8h*)(pr + 40 + half * 16);
#pragma unroll
            for (int e = 0; e < 8; ++e) {
                pa0[e] = c0[e]; pa0[e + 8] = c1[e];
                pa1[e] = c2[e]; pa1[e + 8] = c3[e];
            }
        }
        // ---- PV: B element e -> V row j0 + kk + 16*half + e; V^T row (=col d) contiguous
        {
            const _Float16* vB = VT + (size_t)n * SEQ + j0 + 16 * half;
            acc0 = WMMA_F16(pa0, wmma_b_contig(vB), acc0);
            acc0 = WMMA_F16(pa1, wmma_b_contig(vB + 32), acc0);
            vB += (size_t)16 * SEQ;
            acc1 = WMMA_F16(pa0, wmma_b_contig(vB), acc1);
            acc1 = WMMA_F16(pa1, wmma_b_contig(vB + 32), acc1);
            vB += (size_t)16 * SEQ;
            acc2 = WMMA_F16(pa0, wmma_b_contig(vB), acc2);
            acc2 = WMMA_F16(pa1, wmma_b_contig(vB + 32), acc2);
            vB += (size_t)16 * SEQ;
            acc3 = WMMA_F16(pa0, wmma_b_contig(vB), acc3);
            acc3 = WMMA_F16(pa1, wmma_b_contig(vB + 32), acc3);
        }
    }
    // ---- finalize: divide by l, write [B,S,H*hd] f16 for the output projection
    const int b_ = bh / N_HEADS, h = bh % N_HEADS;
#pragma unroll
    for (int r = 0; r < 8; ++r) {
        const float inv = 1.0f / lrun[r];
        const int m = q0 + r + 8 * half;
        _Float16* o = ab + ((size_t)(b_ * SEQ + m)) * D_MODEL + h * HEAD_DIM + n;
        o[0]  = (_Float16)(acc0[r] * inv);
        o[16] = (_Float16)(acc1[r] * inv);
        o[32] = (_Float16)(acc2[r] * inv);
        o[48] = (_Float16)(acc3[r] * inv);
    }
}

// ===================== Kernel 3: output projection (32x64 per wave) =====================
__global__ __launch_bounds__(256) void k_proj(const _Float16* __restrict__ ab,
                                              const _Float16* __restrict__ WoT,
                                              const float* __restrict__ bo,
                                              float* __restrict__ out) {
    const int lane = threadIdx.x & 31, w = threadIdx.x >> 5;
    const int half = lane >> 4, n = lane & 15;
    const int NT = D_MODEL / 64;               // 8 col tiles of 64
    const int tile = blockIdx.x * 8 + w;
    const int tm = tile / NT, tn = tile % NT;

    const _Float16* arow0 = ab + (size_t)(tm * 32 + n) * D_MODEL;
    const _Float16* arow1 = arow0 + (size_t)16 * D_MODEL;
    const _Float16* wbase = WoT + (size_t)(tn * 64 + n) * D_MODEL + 16 * half;

    v8f acc[2][4];
#pragma unroll
    for (int i = 0; i < 2; ++i)
#pragma unroll
        for (int t = 0; t < 4; ++t) acc[i][t] = (v8f){};

    for (int kk = 0; kk < D_MODEL; kk += 32) {
        const v16h a0 = wmma_a_f16(arow0, kk, half);
        const v16h a1 = wmma_a_f16(arow1, kk, half);
#pragma unroll
        for (int t = 0; t < 4; ++t) {
            const v16h b = wmma_b_contig(wbase + (size_t)(t * 16) * D_MODEL + kk);
            acc[0][t] = WMMA_F16(a0, b, acc[0][t]);
            acc[1][t] = WMMA_F16(a1, b, acc[1][t]);
        }
    }
#pragma unroll
    for (int t = 0; t < 4; ++t) {
        const int col = tn * 64 + t * 16 + n;
        const float bv = bo[col];
#pragma unroll
        for (int i = 0; i < 2; ++i)
#pragma unroll
            for (int r = 0; r < 8; ++r) {
                const int m = tm * 32 + i * 16 + r + 8 * half;
                out[(size_t)m * D_MODEL + col] = acc[i][t][r] + bv;
            }
    }
}

extern "C" void kernel_launch(void* const* d_in, const int* in_sizes, int n_in,
                              void* d_out, int out_size, void* d_ws, size_t ws_size,
                              hipStream_t stream) {
    const float* x    = (const float*)d_in[0];
    const float* Wqkv = (const float*)d_in[1];
    const float* bqkv = (const float*)d_in[2];
    const float* Wo   = (const float*)d_in[3];
    const float* bo   = (const float*)d_in[4];
    float* out = (float*)d_out;

    const size_t per = (size_t)BATCH * N_HEADS * SEQ * HEAD_DIM;  // 4,194,304 elems
    _Float16* qb   = (_Float16*)d_ws;
    _Float16* kb   = qb + per;
    _Float16* vtb  = kb + per;
    _Float16* abuf = vtb + per;
    _Float16* x16  = abuf + (size_t)M_TOTAL * D_MODEL;
    _Float16* wqT  = x16 + (size_t)M_TOTAL * D_MODEL;
    _Float16* woT  = wqT + (size_t)D_MODEL * N_QKV;               // total ~44 MB

    k_cvt   <<<(M_TOTAL * D_MODEL / 4) / 256, 256, 0, stream>>>(x, x16);
    k_cvt_wT<<<(D_MODEL * N_QKV) / 256,       256, 0, stream>>>(Wqkv, wqT, D_MODEL, N_QKV);
    k_cvt_wT<<<(D_MODEL * D_MODEL) / 256,     256, 0, stream>>>(Wo,   woT, D_MODEL, D_MODEL);

    k_qkv <<<(M_TOTAL / 32) * (N_QKV / 64) / 8,   256, 0, stream>>>(x16, wqT, bqkv, qb, kb, vtb);
    k_attn<<<BATCH * N_HEADS * (SEQ / 16) / 8,    256, 0, stream>>>(qb, kb, vtb, abuf);
    k_proj<<<(M_TOTAL / 32) * (D_MODEL / 64) / 8, 256, 0, stream>>>(abuf, woT, bo, out);
}
